// DynamicSNNModel_44667659878466
// MI455X (gfx1250) — matrix operand order repaired and checked by reading, not compiled, not run on hardware
//
#include <hip/hip_runtime.h>

typedef __bf16 bf16_t;
typedef __attribute__((ext_vector_type(16))) __bf16 v16bf;
typedef __attribute__((ext_vector_type(8)))  __bf16 v8bf;
typedef __attribute__((ext_vector_type(8)))  float   v8f;
typedef int v4i __attribute__((ext_vector_type(4)));
typedef __attribute__((address_space(1))) v4i* v4i_gptr;   // global int4*
typedef __attribute__((address_space(3))) v4i* v4i_lptr;   // LDS int4*

#ifndef __has_builtin
#define __has_builtin(x) 0
#endif
#if __has_builtin(__builtin_amdgcn_global_load_async_to_lds_b128)
#define HAVE_ASYNC_LDS 1
#endif

// ---------------------------------------------------------------------------
// Problem constants (fixed by the reference)
// ---------------------------------------------------------------------------
#define T_STEPS 100
#define BATCH   256
#define IN_F    700
#define IN_FP   704     // pad K of layer 0 to a multiple of 64
#define HID     2048
#define OUT_F   20
#define OUT_FP  128     // pad layer-4 N up to one workgroup N-tile

#define BETA_LIF 0.9f
#define TH_LIF   1.0f

// LDS tile geometry: WG tile = 64(M) x 128(N), K chunk = 64, double-buffered.
// Row stride padded to 72 bf16 (144 B) => conflict-free ds_load_b128 fragments.
#define KC     64
#define A_STR  72
#define B_STR  72

// ---------------------------------------------------------------------------
// Utility kernels
// ---------------------------------------------------------------------------
__global__ __launch_bounds__(256) void zero_f32_kernel(float* p, long n) {
  long i = (long)blockIdx.x * blockDim.x + threadIdx.x;
  long stride = (long)gridDim.x * blockDim.x;
  for (; i < n; i += stride) p[i] = 0.0f;
}

// f32 [rows][cols] -> bf16 [rows_pad][cols_pad], zero-filled padding
__global__ __launch_bounds__(256) void convert_pad_bf16_kernel(
    const float* __restrict__ src, bf16_t* __restrict__ dst,
    int rows, int cols, int rows_pad, int cols_pad) {
  long idx = (long)blockIdx.x * blockDim.x + threadIdx.x;
  long total = (long)rows_pad * cols_pad;
  if (idx >= total) return;
  int r = (int)(idx / cols_pad);
  int c = (int)(idx % cols_pad);
  float v = (r < rows && c < cols) ? src[(long)r * cols + c] : 0.0f;
  dst[idx] = (bf16_t)v;
}

// ---------------------------------------------------------------------------
// Scheduling fence: nothing may be reordered across this point.
// ---------------------------------------------------------------------------
__device__ __forceinline__ void sched_fence() {
#if __has_builtin(__builtin_amdgcn_sched_barrier)
  __builtin_amdgcn_sched_barrier(0);
#endif
}

// ---------------------------------------------------------------------------
// Async global -> LDS copy of one 16-byte chunk (per lane).
// ---------------------------------------------------------------------------
__device__ __forceinline__ void copy_b128_to_lds(const bf16_t* g, bf16_t* l) {
#if defined(HAVE_ASYNC_LDS)
  __builtin_amdgcn_global_load_async_to_lds_b128(
      (v4i_gptr)g, (v4i_lptr)l, 0, 0);
#else
  *(v8bf*)l = *(const v8bf*)g;   // register-staged fallback (ds_store_b128)
#endif
}

__device__ __forceinline__ void wait_async_copies() {
#if defined(HAVE_ASYNC_LDS)
#if __has_builtin(__builtin_amdgcn_s_wait_asynccnt)
  __builtin_amdgcn_s_wait_asynccnt(0);
#else
  asm volatile("s_wait_asynccnt 0x0" ::: "memory");
#endif
#endif
}

// Stage one K-chunk: A tile 64x64 (512 b128 tasks), B tile 128x64 (1024 tasks).
__device__ __forceinline__ void stage_tiles(
    const bf16_t* __restrict__ A, int lda,
    const bf16_t* __restrict__ W, int ldw,
    int mwg, int nwg, int k0,
    bf16_t* la, bf16_t* lb, int tid)
{
#pragma unroll
  for (int i = 0; i < 2; ++i) {              // A: 64 rows x 8 segs
    int task = tid + i * 256;
    int row = task >> 3, seg = task & 7;
    copy_b128_to_lds(A + (size_t)(mwg + row) * lda + k0 + seg * 8,
                     la + row * A_STR + seg * 8);
  }
#pragma unroll
  for (int i = 0; i < 4; ++i) {              // B: 128 rows x 8 segs
    int task = tid + i * 256;
    int row = task >> 3, seg = task & 7;
    copy_b128_to_lds(W + (size_t)(nwg + row) * ldw + k0 + seg * 8,
                     lb + row * B_STR + seg * 8);
  }
}

// ---------------------------------------------------------------------------
// Fused GEMM (C = A * W^T + b) + LIF update, LDS double-buffered.
// 8 waves per block arranged 4(M) x 2(N); each wave computes a 16x64 strip
// of C via v_wmma_f32_16x16x32_bf16, fragments fed from LDS.
// Per chunk: preload ALL fragments (20 ds_load_b128), sched_barrier, then
// 8 back-to-back WMMAs -> one s_wait_dscnt per chunk instead of 8.
// ---------------------------------------------------------------------------
template<bool STORE_BF16>
__global__ __launch_bounds__(256) void snn_gemm_lif_kernel(
    const bf16_t* __restrict__ A, int lda,
    const bf16_t* __restrict__ W, int ldw,
    const float* __restrict__ bias, int nbias,
    float* __restrict__ mem, int ldm,
    bf16_t* __restrict__ spk_bf, float* __restrict__ spk_f32,
    int ldo, int nvalid, int K)
{
  __shared__ bf16_t lsA[2][64  * A_STR];
  __shared__ bf16_t lsB[2][128 * B_STR];

  const int tid  = threadIdx.x;
  const int wave = tid >> 5;             // 0..7
  const int lane = tid & 31;
  const int half = lane >> 4;
  const int l16  = lane & 15;

  const int wm  = wave & 3;              // M quadrant
  const int wn  = wave >> 2;             // N half
  const int mwg = blockIdx.x * 64;
  const int nwg = blockIdx.y * 128;

  v8f acc[4] = {};
  const int nch = K / KC;

  stage_tiles(A, lda, W, ldw, mwg, nwg, 0, lsA[0], lsB[0], tid);
  wait_async_copies();
  __syncthreads();

  for (int kc = 0; kc < nch; ++kc) {
    const int cur = kc & 1;
    if (kc + 1 < nch)
      stage_tiles(A, lda, W, ldw, mwg, nwg, (kc + 1) * KC,
                  lsA[cur ^ 1], lsB[cur ^ 1], tid);

    const bf16_t* la  = lsA[cur] + (wm * 16 + l16) * A_STR;
    const bf16_t* lb0 = lsB[cur] + (wn * 64 + l16) * B_STR;

    // Phase 1: preload every fragment of this chunk into registers.
    union F { v16bf v; v8bf h[2]; };
    F a[2], b[2][4];
#pragma unroll
    for (int ks2 = 0; ks2 < 2; ++ks2) {
      const int ks = ks2 * 32;
      a[ks2].h[0] = *(const v8bf*)(la + ks + half * 8);
      a[ks2].h[1] = *(const v8bf*)(la + ks + 16 + half * 8);
#pragma unroll
      for (int j = 0; j < 4; ++j) {
        const bf16_t* bp = lb0 + j * 16 * B_STR + ks + half * 8;
        b[ks2][j].h[0] = *(const v8bf*)(bp);
        b[ks2][j].h[1] = *(const v8bf*)(bp + 16);
      }
    }

    // Keep the scheduler from sinking the DS loads into the WMMA sequence.
    sched_fence();

    // Phase 2: 8 WMMAs back-to-back.
#pragma unroll
    for (int ks2 = 0; ks2 < 2; ++ks2)
#pragma unroll
      for (int j = 0; j < 4; ++j)
        acc[j] = __builtin_amdgcn_wmma_f32_16x16x32_bf16(
            false, a[ks2].v, false, b[ks2][j].v, (short)0, acc[j], false, false);

    // One barrier per chunk: after it, next-iteration staging may overwrite
    // the buffer we just read; our own copies into the other buffer are done.
    if (kc + 1 < nch) wait_async_copies();
    __syncthreads();
  }

  // Epilogue: bias + LIF.  acc[j][r] -> M = mwg+wm*16+half*8+r, N = nwg+wn*64+j*16+l16
#pragma unroll
  for (int j = 0; j < 4; ++j) {
    const int n = nwg + wn * 64 + j * 16 + l16;
    const float bval = (n < nbias) ? bias[n] : 0.0f;
#pragma unroll
    for (int r = 0; r < 8; ++r) {
      const int m = mwg + wm * 16 + half * 8 + r;
      const float cur = acc[j][r] + bval;
      const size_t midx = (size_t)m * ldm + n;
      const float mo = mem[midx];
      const float reset = (mo > TH_LIF) ? TH_LIF : 0.0f;  // prev-step spike, detached
      const float mn = BETA_LIF * mo + cur - reset;
      mem[midx] = mn;
      const float s = (mn > TH_LIF) ? 1.0f : 0.0f;
      if (STORE_BF16) {
        spk_bf[(size_t)m * ldo + n] = (bf16_t)s;
      } else {
        if (n < nvalid) spk_f32[(size_t)m * ldo + n] = s;
      }
    }
  }
}

// ---------------------------------------------------------------------------
// Host launch
// ---------------------------------------------------------------------------
extern "C" void kernel_launch(void* const* d_in, const int* in_sizes, int n_in,
                              void* d_out, int out_size, void* d_ws, size_t ws_size,
                              hipStream_t stream) {
  (void)in_sizes; (void)n_in; (void)out_size; (void)ws_size;

  const float* data = (const float*)d_in[0];
  const float* Wf[5]   = {(const float*)d_in[1], (const float*)d_in[3],
                          (const float*)d_in[5], (const float*)d_in[7],
                          (const float*)d_in[9]};
  const float* bias[5] = {(const float*)d_in[2], (const float*)d_in[4],
                          (const float*)d_in[6], (const float*)d_in[8],
                          (const float*)d_in[10]};
  float* out = (float*)d_out;
  char* ws = (char*)d_ws;

  // ---- workspace layout (all chunk sizes are multiples of 256 bytes) ----
  size_t off = 0;
  bf16_t* wb[5];
  const size_t welems[5] = {(size_t)HID * IN_FP, (size_t)HID * HID,
                            (size_t)HID * HID,   (size_t)HID * HID,
                            (size_t)OUT_FP * HID};
  for (int l = 0; l < 5; ++l) { wb[l] = (bf16_t*)(ws + off); off += welems[l] * 2; }

  bf16_t* dataB = (bf16_t*)(ws + off); off += (size_t)T_STEPS * BATCH * IN_FP * 2;

  float* mem[5];
  for (int l = 0; l < 4; ++l) { mem[l] = (float*)(ws + off); off += (size_t)BATCH * HID * 4; }
  mem[4] = (float*)(ws + off); off += (size_t)BATCH * OUT_FP * 4;

  bf16_t* spkA = (bf16_t*)(ws + off); off += (size_t)BATCH * HID * 2;
  bf16_t* spkB = (bf16_t*)(ws + off); off += (size_t)BATCH * HID * 2;

  // ---- one-time (per launch) conversions: weights + data -> bf16 padded ----
  const int wrows[5] = {HID, HID, HID, HID, OUT_F};
  const int wcols[5] = {IN_F, HID, HID, HID, HID};
  const int wrp[5]   = {HID, HID, HID, HID, OUT_FP};
  const int wcp[5]   = {IN_FP, HID, HID, HID, HID};
  for (int l = 0; l < 5; ++l) {
    long tot = (long)wrp[l] * wcp[l];
    convert_pad_bf16_kernel<<<dim3((unsigned)((tot + 255) / 256)), dim3(256), 0, stream>>>(
        Wf[l], wb[l], wrows[l], wcols[l], wrp[l], wcp[l]);
  }
  {
    long tot = (long)T_STEPS * BATCH * IN_FP;
    convert_pad_bf16_kernel<<<dim3((unsigned)((tot + 255) / 256)), dim3(256), 0, stream>>>(
        data, dataB, T_STEPS * BATCH, IN_F, T_STEPS * BATCH, IN_FP);
  }
  {
    // mem[0..4] are contiguous in the layout above — zero them in one pass
    long nz = 4L * BATCH * HID + (long)BATCH * OUT_FP;
    zero_f32_kernel<<<dim3(1024), dim3(256), 0, stream>>>(mem[0], nz);
  }

  // ---- time loop: 5 fused GEMM+LIF kernels per step, stream-ordered ----
  const dim3 blk(256);
  const dim3 gridH(BATCH / 64, HID / 128);    // (4, 16)
  const dim3 gridO(BATCH / 64, OUT_FP / 128); // (4, 1)

  for (int t = 0; t < T_STEPS; ++t) {
    const bf16_t* a0 = dataB + (size_t)t * BATCH * IN_FP;

    // layer 0: [B,704] x [2048,704]^T
    snn_gemm_lif_kernel<true><<<gridH, blk, 0, stream>>>(
        a0, IN_FP, wb[0], IN_FP, bias[0], HID,
        mem[0], HID, spkA, nullptr, HID, HID, IN_FP);

    // layer 1: spkA -> spkB
    snn_gemm_lif_kernel<true><<<gridH, blk, 0, stream>>>(
        spkA, HID, wb[1], HID, bias[1], HID,
        mem[1], HID, spkB, nullptr, HID, HID, HID);

    // layer 2: spkB -> spkA
    snn_gemm_lif_kernel<true><<<gridH, blk, 0, stream>>>(
        spkB, HID, wb[2], HID, bias[2], HID,
        mem[2], HID, spkA, nullptr, HID, HID, HID);

    // layer 3: spkA -> spkB
    snn_gemm_lif_kernel<true><<<gridH, blk, 0, stream>>>(
        spkA, HID, wb[3], HID, bias[3], HID,
        mem[3], HID, spkB, nullptr, HID, HID, HID);

    // layer 4: spkB -> d_out[t] (f32 spikes, n < 20)
    snn_gemm_lif_kernel<false><<<gridO, blk, 0, stream>>>(
        spkB, HID, wb[4], HID, bias[4], OUT_F,
        mem[4], OUT_FP, nullptr, out + (size_t)t * BATCH * OUT_F,
        OUT_F, OUT_F, HID);
  }
}
